// MultiHeadSelfAttention_61710090109667
// MI455X (gfx1250) — matrix operand order repaired and checked by reading, not compiled, not run on hardware
//
#include <hip/hip_runtime.h>
#include <stdint.h>

// MHSA, causal + padding. bf16 WMMA (v_wmma_f32_16x16x32_bf16), f32 accumulate.
// Pipeline: f32->bf16 convert kernels, TDM/async double-buffered QKV GEMM,
// flash attention with TDM tile loads.
// ws layout: q | k | v ([B,H,S,DH] bf16) | xb ([B*S,E] bf16) | Wb ([3E,E] bf16)

#define B_   8
#define S_   1024
#define E_   1024
#define H_   16
#define DH_  64

typedef __attribute__((ext_vector_type(16))) __bf16 v16bf;
typedef __attribute__((ext_vector_type(8)))  __bf16 bf16x8;
typedef __attribute__((ext_vector_type(8)))  float  v8f;
typedef __attribute__((ext_vector_type(4)))  int    v4i_t;
typedef __attribute__((ext_vector_type(4)))  unsigned int u32x4;
typedef __attribute__((ext_vector_type(8)))  int    i32x8;
typedef __attribute__((ext_vector_type(4)))  int    i32x4;

#if defined(__has_builtin)
#  if __has_builtin(__builtin_amdgcn_global_load_async_to_lds_b128) && \
      __has_builtin(__builtin_amdgcn_s_wait_asynccnt)
#    define HAVE_ASYNC_COPY 1
#  endif
#  if __has_builtin(__builtin_amdgcn_tensor_load_to_lds) && \
      __has_builtin(__builtin_amdgcn_s_wait_tensorcnt)
#    define HAVE_TDM 1
#  endif
#endif

typedef __attribute__((address_space(1))) v4i_t gv4i;   // global v4i
typedef __attribute__((address_space(3))) v4i_t sv4i;   // LDS v4i
typedef __attribute__((address_space(3))) char  lds_char;

__device__ __forceinline__ void cp16_g2s(void* l, const void* g) {
#ifdef HAVE_ASYNC_COPY
  __builtin_amdgcn_global_load_async_to_lds_b128((gv4i*)g, (sv4i*)l, 0, 0);
#else
  *(uint4*)l = *(const uint4*)g;
#endif
}

#ifdef HAVE_TDM
// 2D tile DMA: tile_h rows of tile_w bf16, row stride g_stride (elements),
// LDS padded by pad_amount(code) DWORDs every pad_interval(code) DWORDs.
__device__ __forceinline__ void tdm_load_2d(void* lds, const void* g,
                                            unsigned tile_w, unsigned tile_h,
                                            unsigned g_stride,
                                            unsigned pad_interval_code,
                                            unsigned pad_amount_code) {
  const unsigned long long ga = (unsigned long long)(uintptr_t)g;
  const unsigned la = (unsigned)(unsigned long long)(lds_char*)lds;
  u32x4 g0 = { 1u,                                     // count=1 descriptor
               la,                                     // lds_addr
               (unsigned)ga,                           // global_addr[31:0]
               (unsigned)((ga >> 32) & 0x01FFFFFFu) | 0x80000000u };  // type=2
  const unsigned w0 = (1u << 16)                       // data_size = 2B
                    | (1u << 20)                       // pad_enable
                    | (pad_interval_code << 22)
                    | (pad_amount_code  << 25);
  i32x8 g1 = { (int)w0,
               (int)(tile_w << 16),                    // tensor_dim0 (lo16 @63:48)
               (int)(tile_h << 16),                    // tensor_dim1 (lo16 @95:80)
               (int)(tile_w << 16),                    // tile_dim0   (@127:112)
               (int)tile_h,                            // tile_dim1   (@143:128)
               (int)g_stride,                          // tensor_dim0_stride lo32
               0, 0 };
  i32x4 z4 = { 0, 0, 0, 0 };
  i32x8 z8 = { 0, 0, 0, 0, 0, 0, 0, 0 };
  __builtin_amdgcn_tensor_load_to_lds(g0, g1, z4, z4, z8, 0);
}
#endif

__device__ __forceinline__ v16bf cat8(bf16x8 lo, bf16x8 hi) {
  return __builtin_shufflevector(lo, hi, 0,1,2,3,4,5,6,7,8,9,10,11,12,13,14,15);
}
// A fragment (16x32 bf16, row-major in LDS): half==0 -> K {0..7,16..23}, half==1 -> {8..15,24..31}
__device__ __forceinline__ v16bf frag_a(const __bf16* row, int half) {
  return cat8(*(const bf16x8*)(row + half * 8),
              *(const bf16x8*)(row + 16 + half * 8));
}
// B fragment (32x16 bf16; LDS holds B^T rows: N-major, K contiguous): half==0 -> K {0..7,16..23}, half==1 -> {8..15,24..31}
__device__ __forceinline__ v16bf frag_b(const __bf16* col, int half) {
  return cat8(*(const bf16x8*)(col + half * 8),
              *(const bf16x8*)(col + 16 + half * 8));
}

// ---------------------------------------------------------------------------
// Kernel 0: f32 -> bf16 bulk convert (x and W staged once; halves L2 traffic)
// ---------------------------------------------------------------------------
__global__ void __launch_bounds__(256)
f32_to_bf16_kernel(const float* __restrict__ in, __bf16* __restrict__ outp, int n) {
  const int i = (blockIdx.x * 256 + threadIdx.x) * 8;
  if (i >= n) return;
  float4 a = *(const float4*)(in + i);
  float4 b = *(const float4*)(in + i + 4);
  bf16x8 o = { (__bf16)a.x, (__bf16)a.y, (__bf16)a.z, (__bf16)a.w,
               (__bf16)b.x, (__bf16)b.y, (__bf16)b.z, (__bf16)b.w };
  *(bf16x8*)(outp + i) = o;
}

// ---------------------------------------------------------------------------
// Kernel 1: qkv = x @ W^T + b -> bf16 q/k/v in [B,H,S,DH]; Q scaled 1/sqrt(E).
// TDM (or async) double-buffered 128x128 tiles, K-step 32.
// ---------------------------------------------------------------------------
__global__ void __launch_bounds__(256)
qkv_gemm_kernel(const __bf16* __restrict__ xb, const __bf16* __restrict__ wb,
                const float* __restrict__ bias,
                __bf16* __restrict__ qo, __bf16* __restrict__ ko,
                __bf16* __restrict__ vo)
{
  constexpr int AP = 40;  // 64B row + 16B pad: conflict-free fragment reads
  __shared__ __align__(16) __bf16 As[2][128][AP];
  __shared__ __align__(16) __bf16 Bs[2][128][AP];

  const int tid  = threadIdx.x;
  const int lane = tid & 31;
  const int wave = tid >> 5;
  const int half = lane >> 4;
  const int m16  = lane & 15;
  const int wm   = wave >> 2;   // 2 M-slices of 64
  const int wn   = wave & 3;    // 4 N-slices of 32

  const int m0 = blockIdx.y * 128;
  const int n0 = blockIdx.x * 128;
  const __bf16* xrow = xb + (size_t)m0 * E_;
  const __bf16* wrow = wb + (size_t)n0 * E_;

  const v8f vzero = {0.f,0.f,0.f,0.f,0.f,0.f,0.f,0.f};
  v8f acc[4][2];
#pragma unroll
  for (int mt = 0; mt < 4; ++mt)
#pragma unroll
    for (int nt = 0; nt < 2; ++nt) acc[mt][nt] = vzero;

  auto issue = [&](int bufi, int k0) {
#ifdef HAVE_TDM
    if (wave == 0) {
      tdm_load_2d(&As[bufi][0][0], xrow + k0, 32, 128, E_, 3, 3);
      tdm_load_2d(&Bs[bufi][0][0], wrow + k0, 32, 128, E_, 3, 3);
    }
#else
#pragma unroll
    for (int it = 0; it < 2; ++it) {
      const int idx = tid + it * 256;   // 128 rows x 4 16B-chunks
      const int row = idx >> 2;
      const int c   = (idx & 3) * 8;
      cp16_g2s(&As[bufi][row][c], xrow + (size_t)row * E_ + k0 + c);
      cp16_g2s(&Bs[bufi][row][c], wrow + (size_t)row * E_ + k0 + c);
    }
#endif
  };

  auto compute = [&](int bufi) {
    v16bf af[4], bfr[2];
#pragma unroll
    for (int mt = 0; mt < 4; ++mt)
      af[mt] = frag_a(&As[bufi][wm*64 + mt*16 + m16][0], half);
#pragma unroll
    for (int nt = 0; nt < 2; ++nt)
      bfr[nt] = frag_b(&Bs[bufi][wn*32 + nt*16 + m16][0], half);
#pragma unroll
    for (int mt = 0; mt < 4; ++mt)
#pragma unroll
      for (int nt = 0; nt < 2; ++nt)
        acc[mt][nt] = __builtin_amdgcn_wmma_f32_16x16x32_bf16(
            false, af[mt], false, bfr[nt], (short)0, acc[mt][nt], false, false);
  };

  issue(0, 0);
  for (int k0 = 0; k0 < E_ - 32; k0 += 32) {   // peeled last step: imm waits
    const int buf = (k0 >> 5) & 1;
    issue(buf ^ 1, k0 + 32);
#ifdef HAVE_TDM
    if (wave == 0) __builtin_amdgcn_s_wait_tensorcnt(2);  // prefetch in flight
#elif defined(HAVE_ASYNC_COPY)
    __builtin_amdgcn_s_wait_asynccnt(4);
#endif
    __syncthreads();
    compute(buf);
    __syncthreads();
  }
#ifdef HAVE_TDM
  if (wave == 0) __builtin_amdgcn_s_wait_tensorcnt(0);
#elif defined(HAVE_ASYNC_COPY)
  __builtin_amdgcn_s_wait_asynccnt(0);
#endif
  __syncthreads();
  compute(1);                                   // (E_/32 - 1) & 1 == 1

  // epilogue: bias, fold 1/sqrt(E) into Q, scatter to [B,H,S,DH] bf16
#pragma unroll
  for (int nt = 0; nt < 2; ++nt) {
    const int n     = n0 + wn*32 + nt*16 + m16;
    const int which = n >> 10;          // 0=q 1=k 2=v
    const int rem   = n & 1023;
    const int hh    = rem >> 6;
    const int dd    = rem & 63;
    const float bv  = bias[n];
    const float sc  = (which == 0) ? 0.03125f : 1.0f;
    __bf16* dst = (which == 0) ? qo : ((which == 1) ? ko : vo);
#pragma unroll
    for (int mt = 0; mt < 4; ++mt) {
#pragma unroll
      for (int i = 0; i < 8; ++i) {
        const int m  = m0 + wm*64 + mt*16 + half*8 + i;  // C row = i + 8*half
        const int bb = m >> 10;
        const int ss = m & 1023;
        const float v = (acc[mt][nt][i] + bv) * sc;
        dst[(((size_t)(bb * H_ + hh) * S_) + ss) * DH_ + dd] = (__bf16)v;
      }
    }
  }
}

// ---------------------------------------------------------------------------
// Kernel 2: flash attention, one block per (b, h, 128 q-rows), 8 waves.
// ---------------------------------------------------------------------------
__global__ void __launch_bounds__(256)
attn_kernel(const __bf16* __restrict__ qg, const __bf16* __restrict__ kg,
            const __bf16* __restrict__ vg, const int* __restrict__ lens,
            float* __restrict__ out)
{
  constexpr int AP = 72;  // 128B row + 16B pad: conflict-free fragment reads
  __shared__ __align__(16) __bf16 Qs[128][AP];
  __shared__ __align__(16) __bf16 Ks[64][AP];
  __shared__ __align__(16) __bf16 Vt[64][AP];      // transposed: [d][k]
  __shared__ __align__(16) __bf16 Ps[8][16][AP];   // per-wave P staging

  const int tid  = threadIdx.x;
  const int lane = tid & 31;
  const int wave = tid >> 5;
  const int half = lane >> 4;
  const int m16  = lane & 15;

  const int q0  = blockIdx.x * 128;
  const int h   = blockIdx.y;
  const int b   = blockIdx.z;
  const int len = lens[b];

  const size_t head = ((size_t)(b * H_ + h)) * S_ * DH_;
  const __bf16* qp = qg + head;
  const __bf16* kp = kg + head;
  const __bf16* vp = vg + head;

  // Q tile 128x64 (rows contiguous in [B,H,S,DH])
#ifdef HAVE_TDM
  if (wave == 0)
    tdm_load_2d(&Qs[0][0], qp + (size_t)q0 * DH_, 64, 128, DH_, 4, 3);
  if (wave == 0) __builtin_amdgcn_s_wait_tensorcnt(0);
#else
#pragma unroll
  for (int it = 0; it < 4; ++it) {
    const int idx = tid + it * 256;
    const int row = idx >> 3;
    const int c   = (idx & 7) * 8;
    cp16_g2s(&Qs[row][c], qp + (size_t)(q0 + row) * DH_ + c);
  }
#ifdef HAVE_ASYNC_COPY
  __builtin_amdgcn_s_wait_asynccnt(0);
#endif
#endif
  __syncthreads();

  const v8f vzero = {0.f,0.f,0.f,0.f,0.f,0.f,0.f,0.f};
  float mprev[8], rsum[8];
  v8f o[4];
#pragma unroll
  for (int i = 0; i < 8; ++i) { mprev[i] = -1e30f; rsum[i] = 0.f; }
#pragma unroll
  for (int t = 0; t < 4; ++t) o[t] = vzero;

  const int kend = (q0 < len) ? ((q0 + 128 < len) ? (q0 + 128) : len) : 0;

  for (int j0 = 0; j0 < kend; j0 += 64) {
    // K tile 64x64 (row-major = B^T layout for Q*K^T)
#ifdef HAVE_TDM
    if (wave == 0)
      tdm_load_2d(&Ks[0][0], kp + (size_t)j0 * DH_, 64, 64, DH_, 4, 3);
#else
#pragma unroll
    for (int it = 0; it < 2; ++it) {
      const int idx = tid + it * 256;
      const int row = idx >> 3;
      const int c   = (idx & 7) * 8;
      cp16_g2s(&Ks[row][c], kp + (size_t)(j0 + row) * DH_ + c);
    }
#endif
    // V tile 64x64, transposed into Vt[d][k] (B^T layout for P*V)
#pragma unroll
    for (int it = 0; it < 8; ++it) {
      const int idx = tid + it * 256;
      const int row = idx >> 5;         // k
      const int c2  = (idx & 31) * 2;   // d
      const uint32_t two = *(const uint32_t*)(vp + (size_t)(j0 + row) * DH_ + c2);
      Vt[c2][row]     = ((const __bf16*)&two)[0];
      Vt[c2 + 1][row] = ((const __bf16*)&two)[1];
    }
#ifdef HAVE_TDM
    if (wave == 0) __builtin_amdgcn_s_wait_tensorcnt(0);
#elif defined(HAVE_ASYNC_COPY)
    __builtin_amdgcn_s_wait_asynccnt(0);
#endif
    __syncthreads();

    // scores S = Q K^T  (16 x 64 per wave)
    v16bf aq[2];
#pragma unroll
    for (int c = 0; c < 2; ++c)
      aq[c] = frag_a(&Qs[wave*16 + m16][c*32], half);

    float s[4][8];
#pragma unroll
    for (int t = 0; t < 4; ++t) {
      v8f st = vzero;
#pragma unroll
      for (int c = 0; c < 2; ++c) {
        v16bf bk = frag_b(&Ks[t*16 + m16][c*32], half);
        st = __builtin_amdgcn_wmma_f32_16x16x32_bf16(false, aq[c], false, bk,
                                                     (short)0, st, false, false);
      }
      const int col = j0 + t*16 + m16;
#pragma unroll
      for (int i = 0; i < 8; ++i) {
        const int rowg = q0 + wave*16 + half*8 + i;
        const bool valid = (col <= rowg) && (col < len) && (rowg < len);
        s[t][i] = valid ? st[i] : -__builtin_inff();
      }
    }

    // online softmax: each row lives across the 16 lanes of one half
    float alpha[8];
#pragma unroll
    for (int i = 0; i < 8; ++i) {
      float bm = s[0][i];
#pragma unroll
      for (int t = 1; t < 4; ++t) bm = fmaxf(bm, s[t][i]);
#pragma unroll
      for (int off = 1; off < 16; off <<= 1)
        bm = fmaxf(bm, __shfl_xor(bm, off, 16));
      const float mn = fmaxf(mprev[i], bm);   // stays finite even if all masked
      alpha[i] = __expf(mprev[i] - mn);
      mprev[i] = mn;
      float ls = 0.f;
#pragma unroll
      for (int t = 0; t < 4; ++t) { s[t][i] = __expf(s[t][i] - mn); ls += s[t][i]; }
#pragma unroll
      for (int off = 1; off < 16; off <<= 1)
        ls += __shfl_xor(ls, off, 16);
      rsum[i] = rsum[i] * alpha[i] + ls;
    }
#pragma unroll
    for (int t = 0; t < 4; ++t)
#pragma unroll
      for (int i = 0; i < 8; ++i) o[t][i] *= alpha[i];

    // re-layout P (C layout -> A layout) through per-wave LDS
#pragma unroll
    for (int t = 0; t < 4; ++t)
#pragma unroll
      for (int i = 0; i < 8; ++i)
        Ps[wave][half*8 + i][t*16 + m16] = (__bf16)s[t][i];

    v16bf ap[2];
#pragma unroll
    for (int c = 0; c < 2; ++c)
      ap[c] = frag_a(&Ps[wave][m16][c*32], half);
#pragma unroll
    for (int t = 0; t < 4; ++t)
#pragma unroll
      for (int c = 0; c < 2; ++c) {
        v16bf bv = frag_b(&Vt[t*16 + m16][c*32], half);
        o[t] = __builtin_amdgcn_wmma_f32_16x16x32_bf16(false, ap[c], false, bv,
                                                       (short)0, o[t], false, false);
      }
    __syncthreads();   // before next iteration overwrites Ks/Vt
  }

  float inv[8];
#pragma unroll
  for (int i = 0; i < 8; ++i) inv[i] = (rsum[i] > 0.f) ? (1.f / rsum[i]) : 0.f;
#pragma unroll
  for (int t = 0; t < 4; ++t)
#pragma unroll
    for (int i = 0; i < 8; ++i) {
      const int rowg = q0 + wave*16 + half*8 + i;
      const int d    = t*16 + m16;
      out[((size_t)b * S_ + rowg) * E_ + (size_t)h * DH_ + d] = o[t][i] * inv[i];
    }
}

// ---------------------------------------------------------------------------
extern "C" void kernel_launch(void* const* d_in, const int* in_sizes, int n_in,
                              void* d_out, int out_size, void* d_ws, size_t ws_size,
                              hipStream_t stream) {
  (void)in_sizes; (void)n_in; (void)out_size; (void)ws_size;
  const float* x    = (const float*)d_in[0];
  const int*   lens = (const int*)  d_in[1];
  const float* W    = (const float*)d_in[2];
  const float* bias = (const float*)d_in[3];
  float* out = (float*)d_out;

  const size_t per = (size_t)B_ * H_ * S_ * DH_;     // 8388608
  __bf16* qo = (__bf16*)d_ws;
  __bf16* ko = qo + per;
  __bf16* vo = ko + per;
  __bf16* xb = vo + per;                              // [B*S, E] bf16
  __bf16* wb = xb + (size_t)B_ * S_ * E_;             // [3E, E]  bf16

  const int nx = B_ * S_ * E_;      // 8388608
  const int nw = 3 * E_ * E_;       // 3145728
  f32_to_bf16_kernel<<<nx / 2048, 256, 0, stream>>>(x, xb, nx);
  f32_to_bf16_kernel<<<nw / 2048, 256, 0, stream>>>(W, wb, nw);

  dim3 gGemm(3 * E_ / 128, (B_ * S_) / 128);  // (24, 64)
  qkv_gemm_kernel<<<gGemm, 256, 0, stream>>>(xb, wb, bias, qo, ko, vo);

  dim3 gAttn(S_ / 128, H_, B_);               // (8, 16, 8)
  attn_kernel<<<gAttn, 256, 0, stream>>>(qo, ko, vo, lens, out);
}
